// LSTMNet_80848464380559
// MI455X (gfx1250) — matrix operand order repaired and checked by reading, not compiled, not run on hardware
//
#include <hip/hip_runtime.h>
#include <hip/hip_bf16.h>
#include <math.h>

// ---------------------------------------------------------------------------
// Bidirectional 2-layer LSTM for MI455X (gfx1250, wave32, WMMA).
//   B=32, T=512, D=768, H=384, gates G=4H=1536.
//
// Phase 0: one-time f32->bf16 conversion of x and all weights (they are
//          heavily reused; removes cvt ops + halves bytes in the GEMM loop).
// Phase 1: pre = x @ Wih^T + b  as bf16 WMMA GEMMs, 4 N-tiles per wave
//          (A fragment reused across 4 v_wmma per k-step).
// Phase 2: sequential recurrence, 1 workgroup (24 waves) per direction.
//          Shared state: only bf16 h (24 KB LDS). c and f32 h live in
//          registers (each wave owns a private 16-wide hidden slice).
//          Whh streams as bf16 from the 192 MB L2 every step.
// Workspace: ~266 MB of d_ws.
// ---------------------------------------------------------------------------

#define B_ 32
#define T_ 512
#define D_ 768
#define H_ 384
#define G_ 1536            // 4*H
#define MROWS_ (B_ * T_)   // 16384

typedef __attribute__((ext_vector_type(16))) __bf16 v16bf;
typedef __attribute__((ext_vector_type(8)))  __bf16 v8bf;
typedef __attribute__((ext_vector_type(4)))  __bf16 v4bf;
typedef __attribute__((ext_vector_type(8)))  float  v8f;

__device__ __forceinline__ v8f wmma_bf16(v16bf a, v16bf b, v8f c) {
  // D(16x16,f32) = A(16x32,bf16) x B(32x16,bf16) + C
  return __builtin_amdgcn_wmma_f32_16x16x32_bf16(
      /*neg_a=*/false, a, /*neg_b=*/false, b,
      /*c_mod=*/(short)0, c, /*reuse_a=*/false, /*reuse_b=*/false);
}

__device__ __forceinline__ float sigmoidf_(float x) {
  return 1.0f / (1.0f + __expf(-x));
}
__device__ __forceinline__ float tanhf_(float x) {
  // tanh(x) = 2*sigmoid(2x) - 1  (single v_exp_f32)
  return 2.0f / (1.0f + __expf(-2.0f * x)) - 1.0f;
}

// ---------------------------------------------------------------------------
// f32 -> bf16, 4 elements per thread (float4 load, 8-byte store).
// ---------------------------------------------------------------------------
__global__ void cvt_bf16x4_kernel(const float* __restrict__ in,
                                  __bf16* __restrict__ out, int n4) {
  int i = blockIdx.x * blockDim.x + threadIdx.x;
  if (i < n4) {
    float4 v = ((const float4*)in)[i];
    v4bf o;
    o[0] = (__bf16)v.x; o[1] = (__bf16)v.y;
    o[2] = (__bf16)v.z; o[3] = (__bf16)v.w;
    ((v4bf*)out)[i] = o;
  }
}

// ---------------------------------------------------------------------------
// out[M,N] = A[M,K](bf16) @ W[N,K](bf16)^T + bias[N]
// 256 threads = 8 waves/block; each wave computes a 16x64 block
// (1 M-tile x 4 N-tiles), reusing the A fragment across 4 WMMAs.
//
// A fragment (16x32): lanes 0-15 row M=lane, K k0..+15;
//                     lanes 16-31 row M=lane-16, K k0+16..+31.
// B fragment (32x16): lane n holds W[n, k0 + (lane<16?0:16) + 0..15]
//                     (contiguous 32B of the row-major W).
// C/D: lane<16 -> col, VGPR r -> row m0+r; lanes>=16 -> rows m0+8+r.
// ---------------------------------------------------------------------------
__global__ __launch_bounds__(256)
void gemm_bf16_kernel(const __bf16* __restrict__ A, const __bf16* __restrict__ W,
                      const float* __restrict__ bias, float* __restrict__ out,
                      int M, int N, int K) {
  const int lane = threadIdx.x & 31;
  const int wave = threadIdx.x >> 5;
  const int colGroups = N >> 6;                 // N / 64
  const int task = blockIdx.x * 8 + wave;
  const int tm = task / colGroups;
  const int n0 = (task - tm * colGroups) << 6;  // base of 4 N-tiles
  if (tm * 16 >= M) return;  // wave-uniform (EXEC stays all-ones)

  const int hi  = lane >> 4;
  const int l15 = lane & 15;

  const __bf16* ap = A + (size_t)(tm * 16 + l15) * K;
  const __bf16* wp = W + (size_t)(n0 + l15) * K + hi * 16;  // nt=0 B base

  v8f acc[4];
#pragma unroll
  for (int nt = 0; nt < 4; ++nt) acc[nt] = (v8f){};

  for (int k0 = 0; k0 < K; k0 += 32) {
    v16bf a;
    const v8bf a1 = *(const v8bf*)(ap + k0 + hi * 16);
    const v8bf a2 = *(const v8bf*)(ap + k0 + 8 + hi * 16);
#pragma unroll
    for (int i = 0; i < 8; ++i) { a[i] = a1[i]; a[8 + i] = a2[i]; }
#pragma unroll
    for (int nt = 0; nt < 4; ++nt) {
      const v16bf b = *(const v16bf*)(wp + (size_t)nt * 16 * K + k0);
      acc[nt] = wmma_bf16(a, b, acc[nt]);
    }
  }

  const int crow = tm * 16 + hi * 8;
#pragma unroll
  for (int nt = 0; nt < 4; ++nt) {
    const int col = n0 + nt * 16 + l15;
    const float bv = bias[col];
#pragma unroll
    for (int r = 0; r < 8; ++r)
      out[(size_t)(crow + r) * N + col] = acc[nt][r] + bv;
  }
}

// ---------------------------------------------------------------------------
// LSTM recurrence, one block per direction (blockIdx.x: 0=fwd, 1=bwd).
// 768 threads = 24 waves; wave w owns hidden columns [16w, 16w+16) and
// computes its i/f/g/o gate tiles for both batch halves (tm=0,1).
// Only shared state: bf16 h in LDS (24 KB). c / f32 h in registers.
// outF (f32) and outBf (bf16) are optional (null = skip that store).
// ---------------------------------------------------------------------------
__global__ __launch_bounds__(768)
void lstm_rec_kernel(const float* __restrict__ preF, const float* __restrict__ preB,
                     const __bf16* __restrict__ whhF, const __bf16* __restrict__ whhB,
                     const int* __restrict__ mask,
                     float* __restrict__ outF, __bf16* __restrict__ outBf,
                     int outStride) {
  __shared__ __align__(16) __bf16 h_bf[B_ * H_];  // 24 KB

  const int dir = blockIdx.x;
  const float*  pre    = dir ? preB : preF;
  const __bf16* whh    = dir ? whhB : whhF;
  const int     outOff = dir * H_;

  const int tid  = threadIdx.x;
  const int lane = tid & 31;
  const int wave = tid >> 5;    // 0..23
  const int hi   = lane >> 4;
  const int l15  = lane & 15;
  const int hcol = wave * 16 + l15;    // this lane's hidden column

  // Private cell state: rows (tm*16 + hi*8 + r) x column hcol.
  float cReg[2][8], hReg[2][8];
#pragma unroll
  for (int tm = 0; tm < 2; ++tm)
#pragma unroll
    for (int r = 0; r < 8; ++r) { cReg[tm][r] = 0.0f; hReg[tm][r] = 0.0f; }

  for (int i = tid; i < B_ * H_; i += blockDim.x) h_bf[i] = (__bf16)0.0f;
  __syncthreads();

  for (int t = 0; t < T_; ++t) {
    const int ta = dir ? (T_ - 1 - t) : t;

    // Load pre-activations (input projection) as the C accumulators.
    v8f acc[4][2];
#pragma unroll
    for (int g = 0; g < 4; ++g) {
      const int ncol = g * H_ + hcol;
#pragma unroll
      for (int tm = 0; tm < 2; ++tm) {
        const int rowb = tm * 16 + hi * 8;
#pragma unroll
        for (int r = 0; r < 8; ++r)
          acc[g][tm][r] = pre[((size_t)(rowb + r) * T_ + ta) * G_ + ncol];
      }
    }

    // gates += h @ Whh^T   (K = H = 384, 12 k-steps, 8 v_wmma each)
    for (int k0 = 0; k0 < H_; k0 += 32) {
      v16bf afr[2];
#pragma unroll
      for (int tm = 0; tm < 2; ++tm) {
        const __bf16* hp = &h_bf[(tm * 16 + l15) * H_ + k0];
        const v8bf a1 = *(const v8bf*)(hp + hi * 16);
        const v8bf a2 = *(const v8bf*)(hp + 8 + hi * 16);
#pragma unroll
        for (int i = 0; i < 8; ++i) {
          afr[tm][i]     = a1[i];
          afr[tm][8 + i] = a2[i];
        }
      }
#pragma unroll
      for (int g = 0; g < 4; ++g) {
        const int n = g * H_ + hcol;  // Whh row = gate output column
        const v16bf bfr = *(const v16bf*)(whh + (size_t)n * H_ + k0 + hi * 16);
        acc[g][0] = wmma_bf16(afr[0], bfr, acc[g][0]);
        acc[g][1] = wmma_bf16(afr[1], bfr, acc[g][1]);
      }
    }

    __syncthreads();  // all K-loop reads of h_bf done before overwrite

    // Elementwise LSTM cell update on this wave's private cells.
#pragma unroll
    for (int tm = 0; tm < 2; ++tm) {
      const int rowb = tm * 16 + hi * 8;
#pragma unroll
      for (int r = 0; r < 8; ++r) {
        const int b = rowb + r;
        const float ig = sigmoidf_(acc[0][tm][r]);
        const float fg = sigmoidf_(acc[1][tm][r]);
        const float gg = tanhf_(acc[2][tm][r]);
        const float og = sigmoidf_(acc[3][tm][r]);
        const float c_new = fg * cReg[tm][r] + ig * gg;
        const float h_new = og * tanhf_(c_new);
        const float mk = (float)mask[b * T_ + ta];
        const float hb = mk * h_new + (1.0f - mk) * hReg[tm][r];
        cReg[tm][r] = mk * c_new + (1.0f - mk) * cReg[tm][r];
        hReg[tm][r] = hb;
        h_bf[b * H_ + hcol] = (__bf16)hb;
        const size_t oidx = ((size_t)b * T_ + ta) * outStride + outOff + hcol;
        const float ov = hb * mk;     // reference: hs * mask after concat
        if (outF)  outF[oidx]  = ov;
        if (outBf) outBf[oidx] = (__bf16)ov;
      }
    }
    __syncthreads();  // h_bf writes visible before next step's K loop
  }
}

// ---------------------------------------------------------------------------
extern "C" void kernel_launch(void* const* d_in, const int* in_sizes, int n_in,
                              void* d_out, int out_size, void* d_ws, size_t ws_size,
                              hipStream_t stream) {
  const float* x       = (const float*)d_in[0];
  const int*   mask    = (const int*)d_in[1];
  const float* Wih_l0f = (const float*)d_in[2];
  const float* Whh_l0f = (const float*)d_in[3];
  const float* b_l0f   = (const float*)d_in[4];
  const float* Wih_l0b = (const float*)d_in[5];
  const float* Whh_l0b = (const float*)d_in[6];
  const float* b_l0b   = (const float*)d_in[7];
  const float* Wih_l1f = (const float*)d_in[8];
  const float* Whh_l1f = (const float*)d_in[9];
  const float* b_l1f   = (const float*)d_in[10];
  const float* Wih_l1b = (const float*)d_in[11];
  const float* Whh_l1b = (const float*)d_in[12];
  const float* b_l1b   = (const float*)d_in[13];
  float* out = (float*)d_out;

  // Workspace carve-up (~266 MB total).
  char* ws = (char*)d_ws;
  size_t off = 0;
  auto alloc = [&](size_t bytes) -> void* {
    void* p = ws + off;
    off += (bytes + 255) & ~(size_t)255;
    return p;
  };
  const size_t wihElems = (size_t)G_ * D_;       // 1536*768
  const size_t whhElems = (size_t)G_ * H_;       // 1536*384
  const size_t xElems   = (size_t)MROWS_ * D_;   // 16384*768
  const size_t o0Elems  = (size_t)MROWS_ * 2 * H_;

  __bf16* x_bf    = (__bf16*)alloc(xElems * sizeof(__bf16));    // 25.2 MB
  __bf16* out0_bf = (__bf16*)alloc(o0Elems * sizeof(__bf16));   // 25.2 MB
  __bf16* wih0f   = (__bf16*)alloc(wihElems * sizeof(__bf16));
  __bf16* wih0b   = (__bf16*)alloc(wihElems * sizeof(__bf16));
  __bf16* wih1f   = (__bf16*)alloc(wihElems * sizeof(__bf16));
  __bf16* wih1b   = (__bf16*)alloc(wihElems * sizeof(__bf16));
  __bf16* whh0f   = (__bf16*)alloc(whhElems * sizeof(__bf16));
  __bf16* whh0b   = (__bf16*)alloc(whhElems * sizeof(__bf16));
  __bf16* whh1f   = (__bf16*)alloc(whhElems * sizeof(__bf16));
  __bf16* whh1b   = (__bf16*)alloc(whhElems * sizeof(__bf16));
  float*  preF    = (float*)alloc((size_t)MROWS_ * G_ * sizeof(float)); // 100.7 MB
  float*  preB    = (float*)alloc((size_t)MROWS_ * G_ * sizeof(float)); // 100.7 MB

  // 0) One-time f32 -> bf16 conversions (all sizes divisible by 4).
  auto cvt = [&](const float* src, __bf16* dst, size_t n) {
    const int n4 = (int)(n / 4);
    cvt_bf16x4_kernel<<<(n4 + 255) / 256, 256, 0, stream>>>(src, dst, n4);
  };
  cvt(x, x_bf, xElems);
  cvt(Wih_l0f, wih0f, wihElems);  cvt(Wih_l0b, wih0b, wihElems);
  cvt(Wih_l1f, wih1f, wihElems);  cvt(Wih_l1b, wih1b, wihElems);
  cvt(Whh_l0f, whh0f, whhElems);  cvt(Whh_l0b, whh0b, whhElems);
  cvt(Whh_l1f, whh1f, whhElems);  cvt(Whh_l1b, whh1b, whhElems);

  // (M/16) * (N/64) / 8 waves = 1024 * 24 / 8 = 3072 blocks
  const int gemmBlocks = (MROWS_ / 16) * (G_ / 64) / 8;

  // 1) Layer 0 input projections.
  gemm_bf16_kernel<<<gemmBlocks, 256, 0, stream>>>(x_bf, wih0f, b_l0f, preF,
                                                   MROWS_, G_, D_);
  gemm_bf16_kernel<<<gemmBlocks, 256, 0, stream>>>(x_bf, wih0b, b_l0b, preB,
                                                   MROWS_, G_, D_);
  // 2) Layer 0 recurrence -> bf16 activations only (feeds layer 1 GEMM).
  lstm_rec_kernel<<<2, 768, 0, stream>>>(preF, preB, whh0f, whh0b, mask,
                                         nullptr, out0_bf, 2 * H_);
  // 3) Layer 1 input projections (K = 2H = 768).
  gemm_bf16_kernel<<<gemmBlocks, 256, 0, stream>>>(out0_bf, wih1f, b_l1f, preF,
                                                   MROWS_, G_, 2 * H_);
  gemm_bf16_kernel<<<gemmBlocks, 256, 0, stream>>>(out0_bf, wih1b, b_l1b, preB,
                                                   MROWS_, G_, 2 * H_);
  // 4) Layer 1 recurrence -> final f32 output [B,T,2H].
  lstm_rec_kernel<<<2, 768, 0, stream>>>(preF, preB, whh1f, whh1b, mask,
                                         out, nullptr, 2 * H_);
}
